// GAT_14431090115158
// MI455X (gfx1250) — matrix operand order repaired and checked by reading, not compiled, not run on hardware
//
#include <hip/hip_runtime.h>
#include <hip/hip_bf16.h>

// ---------------------------------------------------------------------------
// GAT 2-layer forward for MI455X (gfx1250, wave32).
// GEMMs via v_wmma_f32_16x16x32_f16 (K=128 -> 4 wmma per 16x16 tile).
// Edge softmax/aggregation via L2-resident f32 atomics (192MB L2 holds all
// accumulators). Self-loops folded into the init kernels (no atomics needed).
// ---------------------------------------------------------------------------

#define F_IN   128
#define HID    16
#define HEADS  8
#define CLS    16
#define NEG_SLOPE 0.2f

typedef __attribute__((ext_vector_type(16))) _Float16 v16h;
typedef __attribute__((ext_vector_type(8)))  float    v8f;

// ---- monotonic float<->uint ordering for atomicMax on floats ---------------
__device__ __forceinline__ unsigned ford(float f) {
  unsigned u = __float_as_uint(f);
  return (u & 0x80000000u) ? ~u : (u | 0x80000000u);
}
__device__ __forceinline__ float forddec(unsigned u) {
  return (u & 0x80000000u) ? __uint_as_float(u ^ 0x80000000u)
                           : __uint_as_float(~u);
}
__device__ __forceinline__ float lrelu(float v) {
  return v > 0.f ? v : NEG_SLOPE * v;
}

// ---------------------------------------------------------------------------
// Pack W (row-major [128 x ncols] f32) into WMMA B-fragment layout (f16).
// Fragment (kc, nc): 32 lanes x 16 f16.  h=lane>=16: elem i = W[kc*32+8h+(i&7)+(i>=8?16:0)][nc*16+(lane&15)]
// ---------------------------------------------------------------------------
__global__ void pack_b_kernel(const float* __restrict__ W, int ncols,
                              _Float16* __restrict__ out, int ncTiles) {
  int idx = blockIdx.x * blockDim.x + threadIdx.x;
  int total = 4 * ncTiles * 32 * 16;
  if (idx >= total) return;
  int i    = idx & 15;
  int lane = (idx >> 4) & 31;
  int frag = idx >> 9;              // kc*ncTiles + nc
  int kc = frag / ncTiles;
  int nc = frag % ncTiles;
  int kk = kc * 32 + ((lane >= 16) ? 8 : 0) + (i & 7) + ((i >= 8) ? 16 : 0);
  int nn = nc * 16 + (lane & 15);
  out[idx] = (_Float16)W[kk * ncols + nn];
}

// ---------------------------------------------------------------------------
// Tiled GEMM: C[ntiles*16 x ncTiles*16] = A[.. x 128] * B(packed f16), f32 acc.
// One wave per 16-row tile; A converted f32->f16 on load.
// ---------------------------------------------------------------------------
__global__ void gemm_wmma_kernel(const float* __restrict__ A, int lda,
                                 const _Float16* __restrict__ Bp,
                                 float* __restrict__ C, int ldc,
                                 int ntiles, int ncTiles) {
  int wave = (blockIdx.x * blockDim.x + threadIdx.x) >> 5;
  int lane = threadIdx.x & 31;
  if (wave >= ntiles) return;
  int row0 = wave * 16;
  bool hi  = lane >= 16;
  int  m   = row0 + (lane & 15);
  const float* arow = A + (size_t)m * lda;

  // Load all 4 K-chunks of the A fragment once (K=128).
  v16h afrag[4];
#pragma unroll
  for (int kc = 0; kc < 4; ++kc) {
    int kb = kc * 32 + (hi ? 8 : 0);
    v16h a;
#pragma unroll
    for (int i = 0; i < 8; ++i) {
      a[i]     = (_Float16)arow[kb + i];
      a[8 + i] = (_Float16)arow[kb + 16 + i];
    }
    afrag[kc] = a;
  }

  for (int nc = 0; nc < ncTiles; ++nc) {
    v8f acc = {};
#pragma unroll
    for (int kc = 0; kc < 4; ++kc) {
      const v16h* bp =
          (const v16h*)(Bp + ((size_t)((kc * ncTiles + nc) * 32 + lane)) * 16);
      v16h b = *bp;
      acc = __builtin_amdgcn_wmma_f32_16x16x32_f16(
          false, afrag[kc], false, b, (short)0, acc, false, false);
    }
    int col   = nc * 16 + (lane & 15);
    int rbase = row0 + (hi ? 8 : 0);
#pragma unroll
    for (int r = 0; r < 8; ++r)
      C[(size_t)(rbase + r) * ldc + col] = acc[r];
  }
}

// ---------------------------------------------------------------------------
// alpha_src / alpha_dst per (node, head):  dot(h[i,hd,:], att[hd,:])
// ---------------------------------------------------------------------------
__global__ void alpha_kernel(const float* __restrict__ H,
                             const float* __restrict__ Asrc,
                             const float* __restrict__ Adst,
                             float* __restrict__ as_, float* __restrict__ ad_,
                             int n, int heads, int C) {
  int idx = blockIdx.x * blockDim.x + threadIdx.x;
  if (idx >= n * heads) return;
  int i = idx / heads, hd = idx % heads;
  const float* hp = H + (size_t)i * heads * C + hd * C;
  float s = 0.f, d = 0.f;
#pragma unroll 4
  for (int c = 0; c < C; ++c) {
    float v = hp[c];
    s += v * Asrc[hd * C + c];
    d += v * Adst[hd * C + c];
  }
  as_[idx] = s;
  ad_[idx] = d;
}

// init segment max with the self-loop edge value
__global__ void init_max_kernel(const float* __restrict__ as_,
                                const float* __restrict__ ad_,
                                unsigned* __restrict__ m, int cnt) {
  int idx = blockIdx.x * blockDim.x + threadIdx.x;
  if (idx >= cnt) return;
  m[idx] = ford(lrelu(as_[idx] + ad_[idx]));
}

__global__ void edge_max_kernel(const int* __restrict__ src,
                                const int* __restrict__ dst,
                                const float* __restrict__ as_,
                                const float* __restrict__ ad_,
                                unsigned* __restrict__ m, int E, int heads) {
  int idx = blockIdx.x * blockDim.x + threadIdx.x;
  if (idx >= E * heads) return;
  int e = idx / heads, hd = idx % heads;
  int s = src[e], d = dst[e];
  float v = lrelu(as_[s * heads + hd] + ad_[d * heads + hd]);
  atomicMax(&m[d * heads + hd], ford(v));
}

// init segment sum with the self-loop exp term
__global__ void init_sum_kernel(const float* __restrict__ as_,
                                const float* __restrict__ ad_,
                                const unsigned* __restrict__ m,
                                float* __restrict__ s_, int cnt) {
  int idx = blockIdx.x * blockDim.x + threadIdx.x;
  if (idx >= cnt) return;
  float e = lrelu(as_[idx] + ad_[idx]);
  s_[idx] = expf(e - forddec(m[idx]));
}

__global__ void edge_sum_kernel(const int* __restrict__ src,
                                const int* __restrict__ dst,
                                const float* __restrict__ as_,
                                const float* __restrict__ ad_,
                                const unsigned* __restrict__ m,
                                float* __restrict__ s_, int E, int heads) {
  int idx = blockIdx.x * blockDim.x + threadIdx.x;
  if (idx >= E * heads) return;
  int e = idx / heads, hd = idx % heads;
  int s = src[e], d = dst[e];
  float v = lrelu(as_[s * heads + hd] + ad_[d * heads + hd]);
  atomicAdd(&s_[d * heads + hd], expf(v - forddec(m[d * heads + hd])));
}

// out[i,hd,c] = h[i,hd,c] * w_self(i,hd) + bias[hd*C+c]   (self-loop message)
__global__ void init_agg_kernel(const float* __restrict__ H,
                                const float* __restrict__ as_,
                                const float* __restrict__ ad_,
                                const unsigned* __restrict__ m,
                                const float* __restrict__ s_,
                                const float* __restrict__ bias,
                                float* __restrict__ out,
                                int n, int heads, int C) {
  int idx = blockIdx.x * blockDim.x + threadIdx.x;
  int HC = heads * C;
  if (idx >= n * HC) return;
  int i = idx / HC, hc = idx % HC, hd = hc / C;
  int ih = i * heads + hd;
  float e = lrelu(as_[ih] + ad_[ih]);
  float w = expf(e - forddec(m[ih])) / (s_[ih] + 1e-16f);
  out[idx] = H[idx] * w + bias[hc];
}

__global__ void edge_agg_kernel(const int* __restrict__ src,
                                const int* __restrict__ dst,
                                const float* __restrict__ H,
                                const float* __restrict__ as_,
                                const float* __restrict__ ad_,
                                const unsigned* __restrict__ m,
                                const float* __restrict__ s_,
                                float* __restrict__ out,
                                int E, int heads, int C) {
  int idx = blockIdx.x * blockDim.x + threadIdx.x;
  int HC = heads * C;
  if (idx >= E * HC) return;
  int e = idx / HC, hc = idx % HC, hd = hc / C;
  int sN = src[e], dN = dst[e];
  float v = lrelu(as_[sN * heads + hd] + ad_[dN * heads + hd]);
  float w = expf(v - forddec(m[dN * heads + hd])) / (s_[dN * heads + hd] + 1e-16f);
  atomicAdd(&out[(size_t)dN * HC + hc], H[(size_t)sN * HC + hc] * w);
}

__global__ void elu_kernel(float* __restrict__ x, int cnt) {
  int idx = blockIdx.x * blockDim.x + threadIdx.x;
  if (idx >= cnt) return;
  float v = x[idx];
  x[idx] = v > 0.f ? v : (expf(v) - 1.f);
}

// ---------------------------------------------------------------------------
extern "C" void kernel_launch(void* const* d_in, const int* in_sizes, int n_in,
                              void* d_out, int out_size, void* d_ws, size_t ws_size,
                              hipStream_t stream) {
  const float* x        = (const float*)d_in[0];
  const int*   edge     = (const int*)d_in[1];
  const float* W1       = (const float*)d_in[2];
  const float* att_src1 = (const float*)d_in[3];
  const float* att_dst1 = (const float*)d_in[4];
  const float* b1       = (const float*)d_in[5];
  const float* W2       = (const float*)d_in[6];
  const float* att_src2 = (const float*)d_in[7];
  const float* att_dst2 = (const float*)d_in[8];
  const float* b2       = (const float*)d_in[9];
  float* out = (float*)d_out;

  const int N = in_sizes[0] / F_IN;      // 50000
  const int E = in_sizes[1] / 2;         // 800000
  const int* esrc = edge;
  const int* edst = edge + E;

  // ---- workspace carve-up (256B aligned) ----------------------------------
  char* ws = (char*)d_ws;
  size_t off = 0;
  auto alloc = [&](size_t bytes) {
    void* p = ws + off;
    off = (off + bytes + 255) & ~(size_t)255;
    return p;
  };
  _Float16* W1p = (_Float16*)alloc((size_t)F_IN * (HEADS * HID) * 2);  // 32KB
  _Float16* W2p = (_Float16*)alloc((size_t)(HEADS * HID) * CLS * 2);   // 4KB
  float*    h1  = (float*)alloc((size_t)N * HEADS * HID * 4);          // 25.6MB
  float*    as1 = (float*)alloc((size_t)N * HEADS * 4);
  float*    ad1 = (float*)alloc((size_t)N * HEADS * 4);
  unsigned* m1  = (unsigned*)alloc((size_t)N * HEADS * 4);
  float*    s1  = (float*)alloc((size_t)N * HEADS * 4);
  float*    o1  = (float*)alloc((size_t)N * HEADS * HID * 4);          // 25.6MB
  float*    h2  = (float*)alloc((size_t)N * CLS * 4);
  float*    as2 = (float*)alloc((size_t)N * 4);
  float*    ad2 = (float*)alloc((size_t)N * 4);
  unsigned* m2  = (unsigned*)alloc((size_t)N * 4);
  float*    s2  = (float*)alloc((size_t)N * 4);
  (void)ws_size; (void)n_in; (void)out_size;

  const int B = 256;
  auto g = [](long total, int blk) { return (int)((total + blk - 1) / blk); };
  const int ntiles = N / 16;   // N = 50000 -> 3125 tiles exactly

  // ================= Layer 1 =================
  pack_b_kernel<<<g(4 * 8 * 32 * 16, B), B, 0, stream>>>(W1, HEADS * HID, W1p, 8);
  gemm_wmma_kernel<<<g((long)ntiles * 32, B), B, 0, stream>>>(
      x, F_IN, W1p, h1, HEADS * HID, ntiles, 8);
  alpha_kernel<<<g((long)N * HEADS, B), B, 0, stream>>>(
      h1, att_src1, att_dst1, as1, ad1, N, HEADS, HID);
  init_max_kernel<<<g((long)N * HEADS, B), B, 0, stream>>>(as1, ad1, m1, N * HEADS);
  edge_max_kernel<<<g((long)E * HEADS, B), B, 0, stream>>>(esrc, edst, as1, ad1, m1, E, HEADS);
  init_sum_kernel<<<g((long)N * HEADS, B), B, 0, stream>>>(as1, ad1, m1, s1, N * HEADS);
  edge_sum_kernel<<<g((long)E * HEADS, B), B, 0, stream>>>(esrc, edst, as1, ad1, m1, s1, E, HEADS);
  init_agg_kernel<<<g((long)N * HEADS * HID, B), B, 0, stream>>>(
      h1, as1, ad1, m1, s1, b1, o1, N, HEADS, HID);
  edge_agg_kernel<<<g((long)E * HEADS * HID, B), B, 0, stream>>>(
      esrc, edst, h1, as1, ad1, m1, s1, o1, E, HEADS, HID);
  elu_kernel<<<g((long)N * HEADS * HID, B), B, 0, stream>>>(o1, N * HEADS * HID);

  // ================= Layer 2 (heads=1, C=16, mean over 1 head == identity) ==
  pack_b_kernel<<<g(4 * 1 * 32 * 16, B), B, 0, stream>>>(W2, CLS, W2p, 1);
  gemm_wmma_kernel<<<g((long)ntiles * 32, B), B, 0, stream>>>(
      o1, HEADS * HID, W2p, h2, CLS, ntiles, 1);
  alpha_kernel<<<g((long)N, B), B, 0, stream>>>(h2, att_src2, att_dst2, as2, ad2, N, 1, CLS);
  init_max_kernel<<<g((long)N, B), B, 0, stream>>>(as2, ad2, m2, N);
  edge_max_kernel<<<g((long)E, B), B, 0, stream>>>(esrc, edst, as2, ad2, m2, E, 1);
  init_sum_kernel<<<g((long)N, B), B, 0, stream>>>(as2, ad2, m2, s2, N);
  edge_sum_kernel<<<g((long)E, B), B, 0, stream>>>(esrc, edst, as2, ad2, m2, s2, E, 1);
  init_agg_kernel<<<g((long)N * CLS, B), B, 0, stream>>>(
      h2, as2, ad2, m2, s2, b2, out, N, 1, CLS);
  edge_agg_kernel<<<g((long)E * CLS, B), B, 0, stream>>>(
      esrc, edst, h2, as2, ad2, m2, s2, out, E, 1, CLS);
}
